// ContextReaderModel_46961172414647
// MI455X (gfx1250) — matrix operand order repaired and checked by reading, not compiled, not run on hardware
//
#include <hip/hip_runtime.h>
#include <math.h>

// ---------------------------------------------------------------- types
typedef __attribute__((ext_vector_type(16))) __bf16 v16bf;
typedef __attribute__((ext_vector_type(8)))  float  v8f;
typedef __attribute__((ext_vector_type(4)))  int    v4i;

union ABFrag { uint4 q[2]; v16bf v; };

#define B_  32
#define T_  4096
#define D_  768
#define HEADS_ 4
#define DK_ 64
#define ROWS 128        // rows of T per workgroup (8 waves x 16)
#define NTILE_T (T_/ROWS)   // 32
#define RPH 776         // LDS row pitch in bf16 halfs (768 + 8 pad -> conflict-free)
#define KCH (D_/32)     // 24 k-chunks of 32
#define SCH 4           // k-chunks staged per buffer
#define SPH (KCH/SCH)   // 6 stages per head
#define FRAG_DW (HEADS_*4*KCH*32*8)   // 98304 dwords per weight matrix

// workspace layout (in floats)
#define OFF_Q    0                  // 8192
#define OFF_FK   8192               // 98304 (as uint)
#define OFF_FV   (8192+FRAG_DW)     // 98304 (as uint)
#define OFF_PART (8192+2*FRAG_DW)   // 32*32*4*66 = 270336

// -------- CDNA5 async global->LDS copy (guarded; fallback: load + ds_store)
#if defined(__HIP_DEVICE_COMPILE__) && __has_builtin(__builtin_amdgcn_global_load_async_to_lds_b128)
#define HAVE_ASYNC 1
#else
#define HAVE_ASYNC 0
#endif

#if HAVE_ASYNC
typedef __attribute__((address_space(1))) v4i GV4;   // global int4
typedef __attribute__((address_space(3))) v4i LV4;   // LDS int4
#endif

__device__ __forceinline__ void copy16(const unsigned* g, unsigned* l) {
#if HAVE_ASYNC
    __builtin_amdgcn_global_load_async_to_lds_b128((GV4*)g, (LV4*)l, 0, 0);
#else
    *(uint4*)l = *(const uint4*)g;
#endif
}

__device__ __forceinline__ void wait_async() {
#if HAVE_ASYNC
#if __has_builtin(__builtin_amdgcn_s_wait_asynccnt)
    __builtin_amdgcn_s_wait_asynccnt(0);
#else
    asm volatile("s_wait_asynccnt 0x0" ::: "memory");
#endif
#endif
}

__device__ __forceinline__ void sched_fence() {
#if __has_builtin(__builtin_amdgcn_sched_barrier)
    __builtin_amdgcn_sched_barrier(0);
#endif
}

__device__ __forceinline__ unsigned f2bf(float x) {
    unsigned u = __float_as_uint(x);
    return ((u + 0x7FFFu + ((u >> 16) & 1u)) >> 16) & 0xFFFFu;
}
__device__ __forceinline__ unsigned pack2bf(float a, float b) {
    return f2bf(a) | (f2bf(b) << 16);
}

// ------------------------------------------------ prep A: swizzle WK/WV -> bf16 B-fragments
// frag layout: flat g = (((h*4+nt)*KCH + kc)*32 + lane)*8 + p  (dwords)
// lane: col = lane&15, kbase = (lane>>4)*16; VGPR p holds K = kbase+2p (lo), +1 (hi)
__global__ __launch_bounds__(256) void swizzle_w_kernel(const float* __restrict__ WK,
                                                        const float* __restrict__ WV,
                                                        unsigned* __restrict__ fragK,
                                                        unsigned* __restrict__ fragV) {
    int gid = blockIdx.x * 256 + threadIdx.x;           // 0 .. 2*FRAG_DW-1
    const float* W  = (gid < FRAG_DW) ? WK : WV;
    unsigned*    F  = (gid < FRAG_DW) ? fragK : fragV;
    int g  = (gid < FRAG_DW) ? gid : gid - FRAG_DW;
    int p  = g & 7;
    int l  = (g >> 3) & 31;
    int t1 = g >> 8;
    int kc = t1 % KCH;
    int t2 = t1 / KCH;
    int nt = t2 & 3;
    int h  = t2 >> 2;
    int n  = (nt << 4) + (l & 15);
    int k  = kc * 32 + ((l >> 4) << 4) + (p << 1);
    const float* wp = W + ((size_t)h * D_ + k) * DK_ + n;
    F[g] = pack2bf(wp[0], wp[DK_]);
}

// ------------------------------------------------ prep B: q[b,h,k] = h_i . WQ
__global__ __launch_bounds__(256) void proj_q_kernel(const float* __restrict__ H,
                                                     const int* __restrict__ npos,
                                                     const float* __restrict__ WQ,
                                                     float* __restrict__ qbuf) {
    int b = blockIdx.x, tid = threadIdx.x;
    int h = tid >> 6, k = tid & 63;
    int np = npos[b];
    const float* hr = H + ((size_t)b * T_ + np) * D_;
    const float* w  = WQ + (size_t)h * D_ * DK_ + k;
    float acc = 0.f;
    for (int d = 0; d < D_; ++d) acc += hr[d] * w[(size_t)d * DK_];
    qbuf[b * (HEADS_ * DK_) + tid] = acc;
}

// stage SCH k-chunks (K+V, 4 n-tiles each = 32KB) into an LDS buffer.
// per (kcl,proj,nt) block of 256 dwords: [0..127] = all lanes' q0 (lane*4),
//                                        [128..255] = all lanes' q1 -> conflict-free ds_load_b128
__device__ __forceinline__ void stage_frag4(const unsigned* __restrict__ fragK,
                                            const unsigned* __restrict__ fragV,
                                            unsigned* sbuf, int h, int ks, int tid) {
#pragma unroll
    for (int r = 0; r < 8; ++r) {
        int c     = tid + (r << 8);       // 0..2047 : 16B chunk id
        int kcl   = c >> 9;               // k-chunk within stage
        int rest  = c & 511;
        int proj  = rest >> 8;            // 0=K, 1=V
        int nt    = (rest >> 6) & 3;
        int i     = rest & 63;            // 16B unit inside 1KB source block
        int lane2 = i >> 1, half = i & 1;
        int kc    = ks * SCH + kcl;
        const unsigned* src = (proj ? fragV : fragK)
            + (((h * 4 + nt) * KCH + kc) * 32) * 8 + i * 4;
        unsigned* dst = sbuf + kcl * 2048 + (proj * 4 + nt) * 256 + half * 128 + lane2 * 4;
        copy16(src, dst);
    }
}

// ------------------------------------------------ main fused attention tile kernel
__global__ __launch_bounds__(256) void attn_main_kernel(const float* __restrict__ H,
                                                        const int* __restrict__ mask,
                                                        const int* __restrict__ npos,
                                                        const float* __restrict__ relb,
                                                        const unsigned* __restrict__ fragK,
                                                        const unsigned* __restrict__ fragV,
                                                        const float* __restrict__ qbuf,
                                                        float* __restrict__ part) {
    __shared__ __align__(16) unsigned short sH[ROWS * RPH];   // 198656 B, bf16 tile
    __shared__ __align__(16) unsigned sFrag[2][SCH * 2048];   // 65536 B, double-buffered weights
    __shared__ float sPart[8 * HEADS_ * 66];                  // 8448 B

    int b    = blockIdx.x >> 5;
    int tile = blockIdx.x & 31;
    int t0   = tile * ROWS;
    int tid  = threadIdx.x;

    // ---- load + convert H tile (fp32 -> bf16) into LDS, coalesced float4
    const float4* Hv = (const float4*)H;
    size_t gbase = ((size_t)b * T_ + t0) * (D_ / 4);
    for (int idx = tid; idx < ROWS * (D_ / 4); idx += 256) {
        int r = idx / (D_ / 4), c4 = idx % (D_ / 4);
        float4 f = Hv[gbase + (size_t)r * (D_ / 4) + c4];
        uint2 u; u.x = pack2bf(f.x, f.y); u.y = pack2bf(f.z, f.w);
        *(uint2*)&sH[r * RPH + c4 * 4] = u;
    }

    int wave  = tid >> 5, lane = tid & 31;
    int rowA  = lane & 15;
    int kb    = (lane >> 4) << 4;      // A-frag K base (halfs): 0 or 16
    int rhalf = (lane >> 4) * 8;       // C/D rows this half holds: j or j+8
    int np    = npos[b];

    // per-row (per this wave's 16 rows) mask & rel-bias delta, hoisted over heads
    int   mk[8], dl[8];
#pragma unroll
    for (int j = 0; j < 8; ++j) {
        int t = t0 + wave * 16 + rhalf + j;
        mk[j] = mask[(size_t)b * T_ + t];
        int d = t - np;
        d = (d < -32) ? -32 : (d > 32 ? 32 : d);
        dl[j] = d + 32;
    }

    // pipeline prologue: stage (h=0,ks=0) into buffer 0
    stage_frag4(fragK, fragV, &sFrag[0][0], 0, 0, tid);

    int s = 0;                                     // global stage index = h*SPH + ks
    for (int h = 0; h < HEADS_; ++h) {
        float qv[4];
#pragma unroll
        for (int nt = 0; nt < 4; ++nt)
            qv[nt] = qbuf[b * (HEADS_ * DK_) + h * DK_ + nt * 16 + (lane & 15)];

        v8f kacc[4] = {{}, {}, {}, {}};
        v8f vacc[4] = {{}, {}, {}, {}};
        for (int ks = 0; ks < SPH; ++ks, ++s) {
            wait_async();                // my stage(s) stores are done
            __syncthreads();             // everyone's stage(s) visible; buf[(s+1)&1] free
            if (s + 1 < HEADS_ * SPH) {
                int s1 = s + 1;
                stage_frag4(fragK, fragV, &sFrag[s1 & 1][0], s1 / SPH, s1 % SPH, tid);
            }
            const unsigned* sbase = &sFrag[s & 1][0];
#pragma unroll
            for (int kcl = 0; kcl < SCH; ++kcl) {
                int kc = ks * SCH + kcl;
                const uint4* sf = (const uint4*)(sbase + kcl * 2048);
                ABFrag a, bk[4], bv[4];
                int abase = rowA * RPH + kc * 32 + kb;
                a.q[0] = *(const uint4*)&sH[abase];
                a.q[1] = *(const uint4*)&sH[abase + 8];
#pragma unroll
                for (int nt = 0; nt < 4; ++nt) {
                    int i0 = nt * 64 + lane;       // uint4 index of this lane's q0
                    bk[nt].q[0] = sf[i0];       bk[nt].q[1] = sf[i0 + 32];
                    bv[nt].q[0] = sf[i0 + 256]; bv[nt].q[1] = sf[i0 + 288];
                }
                sched_fence();   // keep all 16 ds_loads ahead of the WMMA block
#pragma unroll
                for (int nt = 0; nt < 4; ++nt) {
                    kacc[nt] = __builtin_amdgcn_wmma_f32_16x16x32_bf16(
                        false, a.v, false, bk[nt].v, (short)0, kacc[nt], false, false);
                    vacc[nt] = __builtin_amdgcn_wmma_f32_16x16x32_bf16(
                        false, a.v, false, bv[nt].v, (short)0, vacc[nt], false, false);
                }
            }
        }

        // ---- scores for the 16 rows (butterfly over 16-lane halves)
        float sc[8], e[8];
        float mloc = -INFINITY;
#pragma unroll
        for (int j = 0; j < 8; ++j) {
            float p = kacc[0][j] * qv[0] + kacc[1][j] * qv[1] +
                      kacc[2][j] * qv[2] + kacc[3][j] * qv[3];
            p += __shfl_xor(p, 1); p += __shfl_xor(p, 2);
            p += __shfl_xor(p, 4); p += __shfl_xor(p, 8);
            float sv = p * 0.125f + relb[h * 65 + dl[j]];
            if (mk[j] == 0) sv = -1e9f;
            sc[j] = sv;
            mloc  = fmaxf(mloc, sv);
        }
        float M = fmaxf(mloc, __shfl_xor(mloc, 16));
        float ls = 0.f;
#pragma unroll
        for (int j = 0; j < 8; ++j) { e[j] = expf(sc[j] - M); ls += e[j]; }
        float L = ls + __shfl_xor(ls, 16);

        // ---- c_partial[col] = sum_rows e_r * V[r,col]; rows split across lane halves
        int pb = (wave * HEADS_ + h) * 66;
#pragma unroll
        for (int nt = 0; nt < 4; ++nt) {
            float c = 0.f;
#pragma unroll
            for (int j = 0; j < 8; ++j) c += e[j] * vacc[nt][j];
            c += __shfl_xor(c, 16);          // combine rows 0-7 with 8-15
            if (lane < 16) sPart[pb + 2 + nt * 16 + lane] = c;
        }
        if (lane == 0) { sPart[pb] = M; sPart[pb + 1] = L; }
    }
    __syncthreads();

    // ---- combine 8 wave partials -> one tile partial per (b,tile,head)
    int head = tid >> 6, col = tid & 63;
    float M = -INFINITY;
    for (int w = 0; w < 8; ++w)
        M = fmaxf(M, sPart[(w * HEADS_ + head) * 66]);
    float L = 0.f, C = 0.f;
    for (int w = 0; w < 8; ++w) {
        int o = (w * HEADS_ + head) * 66;
        float wt = expf(sPart[o] - M);
        L += wt * sPart[o + 1];
        C += wt * sPart[o + 2 + col];
    }
    size_t po = (((size_t)b * NTILE_T + tile) * HEADS_ + head) * 66;
    if (col == 0) { part[po] = M; part[po + 1] = L; }
    part[po + 2 + col] = C;
}

// ------------------------------------------------ finish: reduce tiles + MLP head
__global__ __launch_bounds__(256) void finish_kernel(const float* __restrict__ H,
                                                     const int* __restrict__ npos,
                                                     const float* __restrict__ pooled,
                                                     const float* __restrict__ part,
                                                     const float* __restrict__ proj_w,
                                                     const float* __restrict__ proj_b,
                                                     const float* __restrict__ mlp_w,
                                                     const float* __restrict__ mlp_b,
                                                     float* __restrict__ out) {
    __shared__ __align__(16) float rv[1792];
    __shared__ float hbuf[256];
    int b = blockIdx.x, tid = threadIdx.x;
    int np = npos[b];
    for (int i = tid; i < D_; i += 256) {
        rv[i]        = H[((size_t)b * T_ + np) * D_ + i];
        rv[1024 + i] = pooled[(size_t)b * D_ + i];
    }
    int head = tid >> 6, col = tid & 63;
    float M = -INFINITY;
    for (int t = 0; t < NTILE_T; ++t)
        M = fmaxf(M, part[(((size_t)b * NTILE_T + t) * HEADS_ + head) * 66]);
    float L = 0.f, C = 0.f;
    for (int t = 0; t < NTILE_T; ++t) {
        size_t o = (((size_t)b * NTILE_T + t) * HEADS_ + head) * 66;
        float wt = expf(part[o] - M);
        L += wt * part[o + 1];
        C += wt * part[o + 2 + col];
    }
    rv[D_ + head * 64 + col] = C / L;
    __syncthreads();

    // hidden = gelu(r @ proj_w.T + proj_b), one output per thread
    float acc = proj_b[tid];
    const float4* wr = (const float4*)(proj_w + (size_t)tid * 1792);
    const float4* r4 = (const float4*)rv;
    for (int i = 0; i < 448; ++i) {
        float4 w4 = wr[i]; float4 x = r4[i];
        acc += w4.x * x.x + w4.y * x.y + w4.z * x.z + w4.w * x.w;
    }
    hbuf[tid] = 0.5f * acc * (1.0f + erff(acc * 0.70710678118654752f));
    __syncthreads();

    if (tid < 64) {
        int o = tid >> 5, l = tid & 31;
        float p = 0.f;
        for (int k = 0; k < 8; ++k) {
            int i = l + k * 32;
            p += hbuf[i] * mlp_w[o * 256 + i];
        }
        for (int m = 16; m >= 1; m >>= 1) p += __shfl_xor(p, m);
        if (l == 0) out[b * 2 + o] = p + mlp_b[o];
    }
}

// ------------------------------------------------ host launcher
extern "C" void kernel_launch(void* const* d_in, const int* in_sizes, int n_in,
                              void* d_out, int out_size, void* d_ws, size_t ws_size,
                              hipStream_t stream) {
    const float* H      = (const float*)d_in[0];
    const int*   mask   = (const int*)  d_in[1];
    const int*   npos   = (const int*)  d_in[2];
    const float* pooled = (const float*)d_in[3];
    const float* WQ     = (const float*)d_in[4];
    const float* WK     = (const float*)d_in[5];
    const float* WV     = (const float*)d_in[6];
    const float* relb   = (const float*)d_in[7];
    const float* projw  = (const float*)d_in[8];
    const float* projb  = (const float*)d_in[9];
    const float* mlpw   = (const float*)d_in[10];
    const float* mlpb   = (const float*)d_in[11];

    float*    ws    = (float*)d_ws;
    float*    qbuf  = ws + OFF_Q;
    unsigned* fragK = (unsigned*)(ws + OFF_FK);
    unsigned* fragV = (unsigned*)(ws + OFF_FV);
    float*    partp = ws + OFF_PART;

    swizzle_w_kernel<<<(2 * FRAG_DW) / 256, 256, 0, stream>>>(WK, WV, fragK, fragV);
    proj_q_kernel<<<B_, 256, 0, stream>>>(H, npos, WQ, qbuf);
    attn_main_kernel<<<B_ * NTILE_T, 256, 0, stream>>>(H, mask, npos, relb,
                                                       fragK, fragV, qbuf, partp);
    finish_kernel<<<B_, 256, 0, stream>>>(H, npos, pooled, partp,
                                          projw, projb, mlpw, mlpb, (float*)d_out);
}
